// GraphNetwork_10651518894531
// MI455X (gfx1250) — matrix operand order repaired and checked by reading, not compiled, not run on hardware
//
#include <hip/hip_runtime.h>
#include <hip/hip_bf16.h>

typedef __attribute__((ext_vector_type(16))) _Float16 v16h;
typedef __attribute__((ext_vector_type(8)))  _Float16 v8h;
typedef __attribute__((ext_vector_type(8)))  float    v8f;

// ---------------------------------------------------------------------------
// Pack fp32 [M,K] row-major -> f16 [M,Kp] row-major, zero-padded (Kp = K
// rounded up to 32). Removes all bounds checks from the GEMM inner loop.
// ---------------------------------------------------------------------------
__global__ void pack_a_f16_kernel(const float* __restrict__ A,
                                  _Float16* __restrict__ Ah,
                                  long long M, int K, int Kp)
{
    const long long i = (long long)blockIdx.x * blockDim.x + threadIdx.x;
    if (i < M * (long long)Kp) {
        const long long r = i / Kp;
        const int c = (int)(i - r * Kp);
        Ah[i] = (c < K) ? (_Float16)A[r * K + c] : (_Float16)0.f;
    }
}

// ---------------------------------------------------------------------------
// Pack fp32 weight B[K,Nn] into WMMA B-fragment layout, f16, zero-padded K:
//   Bp[((kt*ntiles + nt)*32 + lane)*16 + j] = B[kt*32 + (lane>=16?8:0)+(j&7)+(j>=8?16:0)][nt*16+(lane&15)]
// One wave then loads a whole fragment as 32 contiguous bytes per lane.
// ---------------------------------------------------------------------------
__global__ void pack_b_frag_kernel(const float* __restrict__ B,
                                   _Float16* __restrict__ Bp,
                                   int K, int Nn, int Kp)
{
    const int ntiles = Nn / 16;
    const int total = (Kp / 32) * ntiles * 32 * 16;
    const int i = blockIdx.x * blockDim.x + threadIdx.x;
    if (i < total) {
        const int j    = i & 15;
        const int lane = (i >> 4) & 31;
        const int tile = i >> 9;
        const int nt   = tile % ntiles;
        const int kt   = tile / ntiles;
        const int col  = nt * 16 + (lane & 15);
        const int kk   = kt * 32 + ((lane >= 16) ? 8 : 0) + (j & 7) + ((j >= 8) ? 16 : 0);
        Bp[i] = (kk < K) ? (_Float16)B[(size_t)kk * Nn + col] : (_Float16)0.f;
    }
}

// ---------------------------------------------------------------------------
// Fused GEMM: C = relu( sum_i A_i[M,Kp_i] @ B_i + bias ), f16 in, f32 acc.
// A_i: f16 row-major zero-padded; B_i: packed fragment layout (above).
// One wave computes a 16 x (16*TN) strip; no bounds checks, all loads
// unconditional and vectorized. Optionally writes fp32 and/or f16 output.
// WMMA fragment layouts per CDNA5 ISA 7.12.2 (wave32).
// ---------------------------------------------------------------------------
template<int TN>
__global__ __launch_bounds__(256) void gemm3_wmma_packed(
    const _Float16* __restrict__ A0, int Kp0, const _Float16* __restrict__ Bp0,
    const _Float16* __restrict__ A1, int Kp1, const _Float16* __restrict__ Bp1,
    const _Float16* __restrict__ A2, int Kp2, const _Float16* __restrict__ Bp2,
    const float* __restrict__ bias,
    float* __restrict__ Cf,        // fp32 output or nullptr
    _Float16* __restrict__ Ch,     // f16 output or nullptr
    int M, int Nn)
{
    const int lane = threadIdx.x & 31;
    const int wave = blockIdx.x * (blockDim.x >> 5) + (threadIdx.x >> 5);
    const int ntiles_all = Nn / 16;
    const int groups_n = Nn / (16 * TN);
    const int strips_m = M / 16;              // M is a multiple of 16 here
    if (wave >= strips_m * groups_n) return;
    const int tileM  = (wave / groups_n) * 16;
    const int tileN0 = (wave % groups_n) * (16 * TN);
    const int nt0    = tileN0 / 16;

    const bool hi  = (lane >= 16);
    const int  l15 = lane & 15;
    const int  rowA = tileM + l15;

    v8f acc[TN];
    #pragma unroll
    for (int t = 0; t < TN; ++t)
        acc[t] = (v8f){0.f,0.f,0.f,0.f,0.f,0.f,0.f,0.f};

    const _Float16* Aps[3] = {A0, A1, A2};
    const _Float16* Bps[3] = {Bp0, Bp1, Bp2};
    const int       Kps[3] = {Kp0, Kp1, Kp2};

    for (int m = 0; m < 3; ++m) {
        const _Float16* __restrict__ A = Aps[m];
        if (A == nullptr) continue;
        const _Float16* __restrict__ Bp = Bps[m];
        const int Kp = Kps[m];
        const int kts = Kp >> 5;

        const _Float16* __restrict__ arow = A + (size_t)rowA * Kp + (hi ? 8 : 0);

        for (int kt = 0; kt < kts; ++kt) {
            // A fragment: two contiguous 16B loads per lane.
            const v8h alo = *(const v8h*)(arow);
            const v8h ahi = *(const v8h*)(arow + 16);
            const v16h a = __builtin_shufflevector(alo, ahi,
                0,1,2,3,4,5,6,7,8,9,10,11,12,13,14,15);
            arow += 32;

            const _Float16* __restrict__ bbase =
                Bp + (((size_t)kt * ntiles_all + nt0) * 32 + lane) * 16;
            #pragma unroll
            for (int t = 0; t < TN; ++t) {
                const v16h b = *(const v16h*)(bbase + (size_t)t * 512);
                acc[t] = __builtin_amdgcn_wmma_f32_16x16x32_f16(
                    false, a, false, b, (short)0, acc[t], false, false);
            }
        }
    }

    // Store: lanes 0-15 hold rows 0..7 of column l15; lanes 16-31 rows 8..15.
    const int r0 = hi ? 8 : 0;
    #pragma unroll
    for (int t = 0; t < TN; ++t) {
        const int col = tileN0 + t * 16 + l15;
        const float bv = bias[col];
        #pragma unroll
        for (int r = 0; r < 8; ++r) {
            float v = acc[t][r] + bv;
            v = v > 0.f ? v : 0.f;
            const size_t idx = (size_t)(tileM + r0 + r) * Nn + col;
            if (Cf) Cf[idx] = v;
            if (Ch) Ch[idx] = (_Float16)v;
        }
    }
}

// ---------------------------------------------------------------------------
// Count incoming/outgoing edges per node (float counts).
// ---------------------------------------------------------------------------
__global__ void edge_count_kernel(const int* __restrict__ recv,
                                  const int* __restrict__ send,
                                  float* __restrict__ cnt_in,
                                  float* __restrict__ cnt_out, int E)
{
    const int i = blockIdx.x * blockDim.x + threadIdx.x;
    if (i < E) {
        atomicAdd(&cnt_in[recv[i]], 1.f);
        atomicAdd(&cnt_out[send[i]], 1.f);
    }
}

// ---------------------------------------------------------------------------
// Scatter e-rows into per-node sums (receivers and senders) and accumulate
// the global column sum (one atomic per column per block). blockDim.x == H.
// ---------------------------------------------------------------------------
template<typename T>
__global__ void edge_scatter_kernel(const T* __restrict__ e,
                                    const int* __restrict__ recv,
                                    const int* __restrict__ send,
                                    float* __restrict__ inc_sum,
                                    float* __restrict__ out_sum,
                                    float* __restrict__ colsum,
                                    int E, int H)
{
    const int f = threadIdx.x;
    const long long chunk = ((long long)E + gridDim.x - 1) / gridDim.x;
    const long long b0 = (long long)blockIdx.x * chunk;
    const long long b1 = (b0 + chunk < (long long)E) ? b0 + chunk : (long long)E;
    float local = 0.f;
    for (long long i = b0; i < b1; ++i) {
        const float v = (float)e[i * H + f];
        atomicAdd(&inc_sum[(size_t)recv[i] * H + f], v);
        atomicAdd(&out_sum[(size_t)send[i] * H + f], v);
        local += v;
    }
    if (b0 < b1) atomicAdd(&colsum[f], local);
}

// ---------------------------------------------------------------------------
// mean = sum / max(cnt,1), written as f16 (GEMM input form).
// ---------------------------------------------------------------------------
__global__ void seg_normalize_f16_kernel(const float* __restrict__ sum,
                                         const float* __restrict__ cnt,
                                         _Float16* __restrict__ outh,
                                         long long total, int H)
{
    const long long i = (long long)blockIdx.x * blockDim.x + threadIdx.x;
    if (i < total) {
        const float c = cnt[i / H];
        outh[i] = (_Float16)(sum[i] / fmaxf(c, 1.f));
    }
}

// ---------------------------------------------------------------------------
// Column-sum of X[M,H] into colsum[H] (atomic per block per column).
// blockDim.x == H.
// ---------------------------------------------------------------------------
template<typename T>
__global__ void col_sum_kernel(const T* __restrict__ X,
                               float* __restrict__ colsum, int M, int H)
{
    const int f = threadIdx.x;
    const long long chunk = ((long long)M + gridDim.x - 1) / gridDim.x;
    const long long b0 = (long long)blockIdx.x * chunk;
    const long long b1 = (b0 + chunk < (long long)M) ? b0 + chunk : (long long)M;
    float local = 0.f;
    for (long long i = b0; i < b1; ++i) local += (float)X[i * H + f];
    if (b0 < b1) atomicAdd(&colsum[f], local);
}

// ---------------------------------------------------------------------------
// uout[j] = relu( uin@Wu + (nsum/Mn)@Wgn + (esum/Me)@Wge + bu )[j]
// Single block, blockDim.x == H.
// ---------------------------------------------------------------------------
__global__ void global_update_kernel(const float* __restrict__ uin, int Ku,
                                     const float* __restrict__ Wu,
                                     const float* __restrict__ nsum, int Mn, int Kn,
                                     const float* __restrict__ Wgn,
                                     const float* __restrict__ esum, int Me, int Ke,
                                     const float* __restrict__ Wge,
                                     const float* __restrict__ bu,
                                     float* __restrict__ uout, int H)
{
    const int j = threadIdx.x;
    float acc = bu[j];
    for (int k = 0; k < Ku; ++k) acc += uin[k] * Wu[(size_t)k * H + j];
    const float invn = 1.f / (float)Mn;
    for (int k = 0; k < Kn; ++k) acc += (nsum[k] * invn) * Wgn[(size_t)k * H + j];
    const float inve = 1.f / (float)Me;
    for (int k = 0; k < Ke; ++k) acc += (esum[k] * inve) * Wge[(size_t)k * H + j];
    uout[j] = fmaxf(acc, 0.f);
}

// ---------------------------------------------------------------------------
extern "C" void kernel_launch(void* const* d_in, const int* in_sizes, int n_in,
                              void* d_out, int out_size, void* d_ws, size_t ws_size,
                              hipStream_t stream) {
    (void)n_in; (void)out_size; (void)ws_size;

    const float* nodes   = (const float*)d_in[0];
    const float* edges   = (const float*)d_in[1];
    const float* u0      = (const float*)d_in[2];
    const int*   senders = (const int*)d_in[3];
    const int*   recvs   = (const int*)d_in[4];
    const float* We1 = (const float*)d_in[5];  const float* be1 = (const float*)d_in[6];
    const float* Wn1 = (const float*)d_in[7];  const float* Win1 = (const float*)d_in[8];
    const float* Wout1 = (const float*)d_in[9]; const float* bn1 = (const float*)d_in[10];
    const float* Wu1 = (const float*)d_in[11]; const float* Wgn1 = (const float*)d_in[12];
    const float* Wge1 = (const float*)d_in[13]; const float* bu1 = (const float*)d_in[14];
    const float* We2 = (const float*)d_in[15]; const float* be2 = (const float*)d_in[16];
    const float* Wn2 = (const float*)d_in[17]; const float* Win2 = (const float*)d_in[18];
    const float* Wout2 = (const float*)d_in[19]; const float* bn2 = (const float*)d_in[20];
    const float* Wu2 = (const float*)d_in[21]; const float* Wgn2 = (const float*)d_in[22];
    const float* Wge2 = (const float*)d_in[23]; const float* bu2 = (const float*)d_in[24];

    // Derive problem sizes from input element counts.
    const int E  = in_sizes[3];             // senders
    const int FG = in_sizes[2];             // globals_u
    const int FE = in_sizes[1] / E;         // edges
    const int H1 = in_sizes[6];             // be1
    const int H2 = in_sizes[16];            // be2
    const int FN = in_sizes[7] / H1;        // Wn1 is [FN,H1]
    const int Nn = in_sizes[0] / FN;        // nodes
    const int FEp = (FE + 31) & ~31;
    const int FNp = (FN + 31) & ~31;

    // ---- workspace layout (byte-based, 256B aligned regions) ----
    size_t off = 0;
    auto alloc = [&](size_t bytes) -> void* {
        off = (off + 255) & ~(size_t)255;
        void* p = (char*)d_ws + off;
        off += bytes;
        return p;
    };
    _Float16* edges_h = (_Float16*)alloc((size_t)E * FEp * 2);
    _Float16* nodes_h = (_Float16*)alloc((size_t)Nn * FNp * 2);
    _Float16* e1h     = (_Float16*)alloc((size_t)E * H1 * 2);
    _Float16* n1h     = (_Float16*)alloc((size_t)Nn * H1 * 2);
    _Float16* inc1h   = (_Float16*)alloc((size_t)Nn * H1 * 2);
    _Float16* out1h   = (_Float16*)alloc((size_t)Nn * H1 * 2);
    _Float16* inc2h   = (_Float16*)alloc((size_t)Nn * H2 * 2);
    _Float16* out2h   = (_Float16*)alloc((size_t)Nn * H2 * 2);
    auto bpack_sz = [](int Kp, int Nc) { return (size_t)(Kp / 32) * (Nc / 16) * 512 * 2; };
    _Float16* We1p  = (_Float16*)alloc(bpack_sz(FEp, H1));
    _Float16* Wn1p  = (_Float16*)alloc(bpack_sz(FNp, H1));
    _Float16* Win1p = (_Float16*)alloc(bpack_sz(H1, H1));
    _Float16* Wout1p= (_Float16*)alloc(bpack_sz(H1, H1));
    _Float16* We2p  = (_Float16*)alloc(bpack_sz(H1, H2));
    _Float16* Wn2p  = (_Float16*)alloc(bpack_sz(H1, H2));
    _Float16* Win2p = (_Float16*)alloc(bpack_sz(H2, H2));
    _Float16* Wout2p= (_Float16*)alloc(bpack_sz(H2, H2));
    float* u1buf = (float*)alloc((size_t)H1 * 4);
    // zeroed region (fp32 atomic accumulators), contiguous:
    char* zbase = (char*)d_ws + ((off + 255) & ~(size_t)255);
    float* inc1   = (float*)alloc((size_t)Nn * H1 * 4);
    float* out1   = (float*)alloc((size_t)Nn * H1 * 4);
    float* inc2   = (float*)alloc((size_t)Nn * H2 * 4);
    float* out2   = (float*)alloc((size_t)Nn * H2 * 4);
    float* cnt_in = (float*)alloc((size_t)Nn * 4);
    float* cnt_out= (float*)alloc((size_t)Nn * 4);
    float* esum1  = (float*)alloc((size_t)H1 * 4);
    float* nsum1  = (float*)alloc((size_t)H1 * 4);
    float* esum2  = (float*)alloc((size_t)H2 * 4);
    float* nsum2  = (float*)alloc((size_t)H2 * 4);
    const size_t zbytes = (size_t)(((char*)d_ws + off) - zbase);

    float* n2out = (float*)d_out;                           // [Nn, H2]
    float* e2out = (float*)d_out + (size_t)Nn * H2;         // [E, H2]
    float* u2out = (float*)d_out + (size_t)Nn * H2 + (size_t)E * H2;  // [H2]

    hipMemsetAsync(zbase, 0, zbytes, stream);

    const int TPB = 256;
    const int WPB = TPB / 32;
    auto gb = [&](long long n) { return (int)((n + TPB - 1) / TPB); };
    auto gemm_blocks = [&](int M, int Ncols, int tn) {
        const long long waves = (long long)(M / 16) * (Ncols / (16 * tn));
        return (int)((waves + WPB - 1) / WPB);
    };
    auto pack_b = [&](const float* B, _Float16* Bp, int K, int Nc, int Kp) {
        const int total = (Kp / 32) * (Nc / 16) * 512;
        pack_b_frag_kernel<<<gb(total), TPB, 0, stream>>>(B, Bp, K, Nc, Kp);
    };

    // ---- operand packing (weights tiny; A operands one streaming pass) ----
    pack_a_f16_kernel<<<gb((long long)E * FEp), TPB, 0, stream>>>(edges, edges_h, E, FE, FEp);
    pack_a_f16_kernel<<<gb((long long)Nn * FNp), TPB, 0, stream>>>(nodes, nodes_h, Nn, FN, FNp);
    pack_b(We1, We1p, FE, H1, FEp);
    pack_b(Wn1, Wn1p, FN, H1, FNp);
    pack_b(Win1, Win1p, H1, H1, H1);
    pack_b(Wout1, Wout1p, H1, H1, H1);
    pack_b(We2, We2p, H1, H2, H1);
    pack_b(Wn2, Wn2p, H1, H2, H1);
    pack_b(Win2, Win2p, H2, H2, H2);
    pack_b(Wout2, Wout2p, H2, H2, H2);

    // per-node edge counts (shared by both blocks)
    edge_count_kernel<<<gb(E), TPB, 0, stream>>>(recvs, senders, cnt_in, cnt_out, E);

    // ---- block 1 ----
    // e1 = relu(edges @ We1 + be1)  -> f16 only
    gemm3_wmma_packed<8><<<gemm_blocks(E, H1, 8), TPB, 0, stream>>>(
        edges_h, FEp, We1p, nullptr, 0, nullptr, nullptr, 0, nullptr,
        be1, nullptr, e1h, E, H1);

    edge_scatter_kernel<_Float16><<<4096, H1, 0, stream>>>(
        e1h, recvs, senders, inc1, out1, esum1, E, H1);

    {
        const long long tot = (long long)Nn * H1;
        seg_normalize_f16_kernel<<<gb(tot), TPB, 0, stream>>>(inc1, cnt_in, inc1h, tot, H1);
        seg_normalize_f16_kernel<<<gb(tot), TPB, 0, stream>>>(out1, cnt_out, out1h, tot, H1);
    }

    // n1 = relu(nodes@Wn1 + inc1@Win1 + out1@Wout1 + bn1) -> f16 only
    gemm3_wmma_packed<8><<<gemm_blocks(Nn, H1, 8), TPB, 0, stream>>>(
        nodes_h, FNp, Wn1p, inc1h, H1, Win1p, out1h, H1, Wout1p,
        bn1, nullptr, n1h, Nn, H1);

    col_sum_kernel<_Float16><<<1024, H1, 0, stream>>>(n1h, nsum1, Nn, H1);

    global_update_kernel<<<1, H1, 0, stream>>>(
        u0, FG, Wu1, nsum1, Nn, H1, Wgn1, esum1, E, H1, Wge1, bu1, u1buf, H1);

    // ---- block 2 ----
    // e2 = relu(e1 @ We2 + be2) -> fp32 to d_out
    gemm3_wmma_packed<8><<<gemm_blocks(E, H2, 8), TPB, 0, stream>>>(
        e1h, H1, We2p, nullptr, 0, nullptr, nullptr, 0, nullptr,
        be2, e2out, nullptr, E, H2);

    edge_scatter_kernel<float><<<4096, H2, 0, stream>>>(
        e2out, recvs, senders, inc2, out2, esum2, E, H2);

    {
        const long long tot = (long long)Nn * H2;
        seg_normalize_f16_kernel<<<gb(tot), TPB, 0, stream>>>(inc2, cnt_in, inc2h, tot, H2);
        seg_normalize_f16_kernel<<<gb(tot), TPB, 0, stream>>>(out2, cnt_out, out2h, tot, H2);
    }

    // n2 = relu(n1@Wn2 + inc2@Win2 + out2@Wout2 + bn2) -> fp32 to d_out
    gemm3_wmma_packed<8><<<gemm_blocks(Nn, H2, 8), TPB, 0, stream>>>(
        n1h, H1, Wn2p, inc2h, H2, Win2p, out2h, H2, Wout2p,
        bn2, n2out, nullptr, Nn, H2);

    col_sum_kernel<float><<<1024, H2, 0, stream>>>(n2out, nsum2, Nn, H2);

    global_update_kernel<<<1, H2, 0, stream>>>(
        u1buf, H1, Wu2, nsum2, Nn, H2, Wgn2, esum2, E, H2, Wge2, bu2, u2out, H2);
}
